// SpatioTemporalCrossAttention_2619930050613
// MI455X (gfx1250) — matrix operand order repaired and checked by reading, not compiled, not run on hardware
//
#include <hip/hip_runtime.h>

// SpatioTemporalCrossAttention for MI455X (gfx1250): f16 WMMA everywhere,
// fp32 accumulation, flash-attention inner loop, double-buffered async
// global->LDS staging, ds_swizzle reductions, fused bias / residual+LN.

#define E    1024
#define H    16
#define D    64
#define SEQ  2048
#define BATCH 2
#define MTOT (BATCH * SEQ) // 4096

typedef __attribute__((ext_vector_type(16))) _Float16 v16h;
typedef __attribute__((ext_vector_type(8)))  _Float16 v8h;
typedef __attribute__((ext_vector_type(4)))  _Float16 v4h;
typedef __attribute__((ext_vector_type(8)))  float    v8f;

union Frag16 { v16h v; v8h h[2]; };

__device__ __forceinline__ v8f wmma16(v16h a, v16h b, v8f c) {
  // D = A(16x32 f16) * B(32x16 f16) + C(16x16 f32)
  return __builtin_amdgcn_wmma_f32_16x16x32_f16(
      /*neg_a=*/false, a, /*neg_b=*/false, b,
      /*c_mod=*/(short)0, c, /*reuse_a=*/false, /*reuse_b=*/false);
}

// gfx1250 async global->LDS copy (ASYNCcnt); 16 bytes per lane.
__device__ __forceinline__ void async_ld_b128(uint32_t lds_byte_off,
                                              const void* gaddr) {
  asm volatile("global_load_async_to_lds_b128 %0, %1, off"
               :: "v"(lds_byte_off), "v"(gaddr) : "memory");
}
__device__ __forceinline__ void wait_async0() {
  asm volatile("s_wait_asynccnt 0x0" ::: "memory");
}
__device__ __forceinline__ uint32_t lds_off(const void* p) {
  return (uint32_t)(uintptr_t)p; // LDS aperture: addr[31:0] is the LDS offset
}

// single-instruction butterfly: ds_swizzle_b32 group-of-32 xor mode
template <int MASK>
__device__ __forceinline__ float swz(float x) {
  return __int_as_float(
      __builtin_amdgcn_ds_swizzle(__float_as_int(x), (MASK << 10) | 0x1f));
}

// ---------------------------------------------------------------- convert
__global__ void cvt_f32_to_f16(const float* __restrict__ in,
                               _Float16* __restrict__ out, int n4) {
  int i = blockIdx.x * blockDim.x + threadIdx.x;
  const int stride = gridDim.x * blockDim.x;
  const float4* in4 = (const float4*)in;
  v4h* out4 = (v4h*)out;
  for (; i < n4; i += stride) {
    float4 v = in4[i];
    v4h o;
    o.x = (_Float16)v.x; o.y = (_Float16)v.y;
    o.z = (_Float16)v.z; o.w = (_Float16)v.w;
    out4[i] = o;
  }
}

// ---------------------------------------------------------------- GEMM
// C(MxN) = A(MxK,f16,row) * B(KxN,f16,row) + bias ; OUT_HALF: store f16 else f32
// Block tile 128x128, BK=32; 8 waves in 2x4 grid; wave tile 64x32 -> 8 WMMAs/step.
// Double-buffered: A via async global->LDS, B via regs -> transposed LDS.
template <bool OUT_HALF>
__global__ void __launch_bounds__(256)
gemm_bias_kernel(const _Float16* __restrict__ A, const _Float16* __restrict__ Bw,
                 const float* __restrict__ bias, void* __restrict__ Cout,
                 int M, int Nn, int K) {
  __shared__ __align__(32) _Float16 As[2][128 * 32]; // 2 x 8 KB, row-major
  __shared__ __align__(32) _Float16 Bt[2][128 * 32]; // 2 x 8 KB, Bt[n][k]

  const int tid  = threadIdx.x;
  const int lane = tid & 31;
  const int wave = tid >> 5;
  const int wr   = wave >> 2; // 2 wave-rows of 64
  const int wc   = wave & 3;  // 4 wave-cols of 32
  const int m0   = blockIdx.y * 128;
  const int n0   = blockIdx.x * 128;
  const int hl   = lane >> 4;
  const int l16  = lane & 15;
  const int hl8  = hl * 8;
  const int kk16 = hl * 16;

  v8f acc[4][2];
#pragma unroll
  for (int i = 0; i < 4; i++)
#pragma unroll
    for (int j = 0; j < 2; j++) { v8f z = {}; acc[i][j] = z; }

  const int arow = tid >> 1;         // 0..127
  const int acol = (tid & 1) * 16;   // 0/16
  const int bn   = tid & 127;        // 0..127
  const int bk0  = (tid >> 7) * 16;  // 0/16
  const uint32_t aoff[2] = { lds_off(&As[0][arow * 32 + acol]),
                             lds_off(&As[1][arow * 32 + acol]) };

  auto issueA = [&](int buf, int k0) {
    const _Float16* ag = &A[(size_t)(m0 + arow) * K + k0 + acol];
    async_ld_b128(aoff[buf],      ag);
    async_ld_b128(aoff[buf] + 16, ag + 8);
  };

  // ---- preload buffer 0
  issueA(0, 0);
#pragma unroll
  for (int i = 0; i < 16; i++)
    Bt[0][bn * 32 + bk0 + i] = Bw[(size_t)(bk0 + i) * Nn + n0 + bn];
  wait_async0();
  __syncthreads();

  for (int k0 = 0; k0 < K; k0 += 32) {
    const int cur = (k0 >> 5) & 1;
    const int nxt = cur ^ 1;
    const bool hasNext = (k0 + 32) < K;

    _Float16 gv[16];
    if (hasNext) {
      issueA(nxt, k0 + 32);
#pragma unroll
      for (int i = 0; i < 16; i++)
        gv[i] = Bw[(size_t)(k0 + 32 + bk0 + i) * Nn + n0 + bn];
    }

    // ---- compute from current buffer
    Frag16 af[4], bf[2];
#pragma unroll
    for (int i = 0; i < 4; i++) {
      const int m = wr * 64 + i * 16 + l16;
      af[i].h[0] = *(const v8h*)&As[cur][m * 32 + kk16];
      af[i].h[1] = *(const v8h*)&As[cur][m * 32 + kk16 + 8];
    }
#pragma unroll
    for (int j = 0; j < 2; j++) {
      const int n = wc * 32 + j * 16 + l16;
      bf[j].v = *(const v16h*)&Bt[cur][n * 32 + kk16];
    }
#pragma unroll
    for (int i = 0; i < 4; i++)
#pragma unroll
      for (int j = 0; j < 2; j++)
        acc[i][j] = wmma16(af[i].v, bf[j].v, acc[i][j]);

    // ---- land next B tile, publish
    if (hasNext) {
#pragma unroll
      for (int i = 0; i < 16; i++)
        Bt[nxt][bn * 32 + bk0 + i] = gv[i];
    }
    wait_async0();
    __syncthreads();
  }

#pragma unroll
  for (int i = 0; i < 4; i++) {
#pragma unroll
    for (int j = 0; j < 2; j++) {
      const int ng = n0 + wc * 32 + j * 16 + l16;
      const float bv = bias[ng];
#pragma unroll
      for (int r = 0; r < 8; r++) {
        const int mg = m0 + wr * 64 + i * 16 + r + 8 * hl;
        const float val = acc[i][j][r] + bv;
        if (OUT_HALF)
          ((_Float16*)Cout)[(size_t)mg * Nn + ng] = (_Float16)val;
        else
          ((float*)Cout)[(size_t)mg * Nn + ng] = val;
      }
    }
  }
}

// ---------------------------------------------------------------- attention
// Q/K/V layout: (B*SEQ) x E, head h occupies columns [h*64, h*64+64)
__global__ void __launch_bounds__(128)
attn_kernel(const _Float16* __restrict__ Qh, const _Float16* __restrict__ Kh,
            const _Float16* __restrict__ Vh, _Float16* __restrict__ Oh) {
  __shared__ __align__(32) _Float16 Ks[2][32 * 64];  // 2 x 4 KB, key-major
  __shared__ __align__(32) _Float16 Vt[2][64 * 32];  // 2 x 4 KB, d-major
  __shared__ __align__(32) _Float16 Ps[4 * 16 * 32]; // per-wave P scratch

  const int tid  = threadIdx.x;
  const int lane = tid & 31;
  const int wave = tid >> 5;
  const int b = blockIdx.z, h = blockIdx.y;
  const int q0 = blockIdx.x * 64 + wave * 16;
  const int hl = lane >> 4, l16 = lane & 15, hl8 = hl * 8, kk16 = hl * 16;

  const size_t headoff = (size_t)h * D;
  const _Float16* qrow = Qh + ((size_t)(b * SEQ + q0 + l16)) * E + headoff;

  // Q fragments for the two d-chunks (contiguous k order, k = 16*hl + s, matching the K frags)
  Frag16 qa[2];
#pragma unroll
  for (int c = 0; c < 2; c++) {
    qa[c].h[0] = *(const v8h*)&qrow[c * 32 + kk16];
    qa[c].h[1] = *(const v8h*)&qrow[c * 32 + kk16 + 8];
  }

  float mrow[8], lrow[8];
  v8f o[4];
#pragma unroll
  for (int r = 0; r < 8; r++) { mrow[r] = -3.0e38f; lrow[r] = 0.0f; }
#pragma unroll
  for (int t = 0; t < 4; t++) { v8f z = {}; o[t] = z; }

  const int skk = tid >> 2;        // 0..31 key row
  const int scb = (tid & 3) * 16;  // 0/16/32/48
  const int vd  = tid & 63;        // 0..63
  const int vk0 = (tid >> 6) * 16; // 0/16
  _Float16* pp = &Ps[wave * 512];
  const uint32_t koff[2] = { lds_off(&Ks[0][skk * 64 + scb]),
                             lds_off(&Ks[1][skk * 64 + scb]) };

  auto issueK = [&](int buf, int kc) {
    const _Float16* kg =
        &Kh[((size_t)(b * SEQ + kc + skk)) * E + headoff + scb];
    async_ld_b128(koff[buf],      kg);
    async_ld_b128(koff[buf] + 16, kg + 8);
  };

  // ---- preload chunk 0
  issueK(0, 0);
#pragma unroll
  for (int i = 0; i < 16; i++)
    Vt[0][vd * 32 + vk0 + i] =
        Vh[((size_t)(b * SEQ + vk0 + i)) * E + headoff + vd];
  wait_async0();
  __syncthreads();

  for (int kc = 0; kc < SEQ; kc += 32) {
    const int cur = (kc >> 5) & 1;
    const int nxt = cur ^ 1;
    const bool hasNext = (kc + 32) < SEQ;

    _Float16 gv[16];
    if (hasNext) {
      issueK(nxt, kc + 32);
#pragma unroll
      for (int i = 0; i < 16; i++)
        gv[i] = Vh[((size_t)(b * SEQ + kc + 32 + vk0 + i)) * E + headoff + vd];
    }

    // S = Q * K^T : two 16x16 tiles (keys 0-15 / 16-31 of chunk)
    v8f s[2];
#pragma unroll
    for (int j = 0; j < 2; j++) {
      v8f z = {}; s[j] = z;
      const int n = j * 16 + l16;
#pragma unroll
      for (int c = 0; c < 2; c++) {
        Frag16 kb;
        kb.v = *(const v16h*)&Ks[cur][n * 64 + c * 32 + kk16];
        s[j] = wmma16(qa[c].v, kb.v, s[j]);
      }
    }

    // online softmax; row m = r + 8*hl lives across a 16-lane half
    const float scale = 0.125f; // 1/sqrt(64)
#pragma unroll
    for (int r = 0; r < 8; r++) {
      float s0 = s[0][r] * scale, s1 = s[1][r] * scale;
      float mx = fmaxf(s0, s1);
      mx = fmaxf(mx, swz<1>(mx));
      mx = fmaxf(mx, swz<2>(mx));
      mx = fmaxf(mx, swz<4>(mx));
      mx = fmaxf(mx, swz<8>(mx));
      const float mnew  = fmaxf(mrow[r], mx);
      const float alpha = __expf(mrow[r] - mnew);
      const float p0 = __expf(s0 - mnew);
      const float p1 = __expf(s1 - mnew);
      float rs = p0 + p1;
      rs += swz<1>(rs);
      rs += swz<2>(rs);
      rs += swz<4>(rs);
      rs += swz<8>(rs);
      lrow[r] = lrow[r] * alpha + rs;
      mrow[r] = mnew;
#pragma unroll
      for (int t = 0; t < 4; t++) o[t][r] = o[t][r] * alpha;
      const int pm = r + 8 * hl;
      pp[pm * 32 + l16]      = (_Float16)p0;
      pp[pm * 32 + 16 + l16] = (_Float16)p1;
    }

    // O += P * V  (re-shape P through wave-private LDS into A-frag layout)
    Frag16 pa;
    pa.h[0] = *(const v8h*)&pp[l16 * 32 + kk16];
    pa.h[1] = *(const v8h*)&pp[l16 * 32 + kk16 + 8];
#pragma unroll
    for (int t = 0; t < 4; t++) {
      Frag16 vb;
      vb.v = *(const v16h*)&Vt[cur][(t * 16 + l16) * 32 + kk16];
      o[t] = wmma16(pa.v, vb.v, o[t]);
    }

    // ---- land next V tile (transposed), publish
    if (hasNext) {
#pragma unroll
      for (int i = 0; i < 16; i++)
        Vt[nxt][vd * 32 + vk0 + i] = gv[i];
    }
    wait_async0();
    __syncthreads();
  }

  // O / l -> f16 output
  _Float16* orow = Oh + ((size_t)(b * SEQ + q0)) * E + headoff;
#pragma unroll
  for (int r = 0; r < 8; r++) {
    const float inv = 1.0f / lrow[r];
    const int m = r + 8 * hl;
#pragma unroll
    for (int t = 0; t < 4; t++)
      orow[(size_t)m * E + t * 16 + l16] = (_Float16)(o[t][r] * inv);
  }
}

// ---------------------------------------------------------------- residual + LN
__global__ void __launch_bounds__(256)
add_ln_kernel(const float* __restrict__ X, const float* __restrict__ Y,
              const float* __restrict__ g, const float* __restrict__ bta,
              float* __restrict__ out) {
  __shared__ float red[2][8];
  const int row = blockIdx.x;
  const int tid = threadIdx.x;
  const size_t base = (size_t)row * E;

  float z[4], s = 0.f, ss = 0.f;
#pragma unroll
  for (int i = 0; i < 4; i++) {
    const int c = tid + i * 256;
    const float v = X[base + c] + Y[base + c];
    z[i] = v; s += v; ss += v * v;
  }
  s += swz<1>(s);  ss += swz<1>(ss);
  s += swz<2>(s);  ss += swz<2>(ss);
  s += swz<4>(s);  ss += swz<4>(ss);
  s += swz<8>(s);  ss += swz<8>(ss);
  s += swz<16>(s); ss += swz<16>(ss);
  if ((tid & 31) == 0) { red[0][tid >> 5] = s; red[1][tid >> 5] = ss; }
  __syncthreads();
  float tot = 0.f, tot2 = 0.f;
#pragma unroll
  for (int k = 0; k < 8; k++) { tot += red[0][k]; tot2 += red[1][k]; }
  const float mu = tot * (1.0f / E);
  float var = tot2 * (1.0f / E) - mu * mu;
  var = fmaxf(var, 0.0f);
  const float rs = rsqrtf(var + 1e-5f);
#pragma unroll
  for (int i = 0; i < 4; i++) {
    const int c = tid + i * 256;
    out[base + c] = (z[i] - mu) * rs * g[c] + bta[c];
  }
}

// ---------------------------------------------------------------- launch
extern "C" void kernel_launch(void* const* d_in, const int* in_sizes, int n_in,
                              void* d_out, int out_size, void* d_ws, size_t ws_size,
                              hipStream_t stream) {
  (void)in_sizes; (void)n_in; (void)out_size; (void)ws_size;
  const float* temp = (const float*)d_in[0];
  const float* spat = (const float*)d_in[1];
  const float* Wf[8];
  const float* bf[8];
  for (int i = 0; i < 8; i++) {          // 0..3 t2s q,k,v,o ; 4..7 s2t q,k,v,o
    Wf[i] = (const float*)d_in[2 + 2 * i];
    bf[i] = (const float*)d_in[3 + 2 * i];
  }
  const float* ln_g = (const float*)d_in[18];
  const float* ln_b = (const float*)d_in[19];
  float* out = (float*)d_out;

  char* w = (char*)d_ws;
  const size_t MB = 1024u * 1024u;
  _Float16* temp_h = (_Float16*)(w + 0 * MB);   // 8 MB
  _Float16* spat_h = (_Float16*)(w + 8 * MB);   // 8 MB
  _Float16* Wh[8];
  for (int i = 0; i < 8; i++) Wh[i] = (_Float16*)(w + 16 * MB + (size_t)i * 2 * MB);
  _Float16* Qh = (_Float16*)(w + 32 * MB);
  _Float16* Kh = (_Float16*)(w + 40 * MB);
  _Float16* Vh = (_Float16*)(w + 48 * MB);
  _Float16* Oh = (_Float16*)(w + 56 * MB);
  float*    Yb = (float*)(w + 64 * MB);          // 16 MB fp32

  cvt_f32_to_f16<<<1024, 256, 0, stream>>>(temp, temp_h, MTOT * E / 4);
  cvt_f32_to_f16<<<1024, 256, 0, stream>>>(spat, spat_h, MTOT * E / 4);
  for (int i = 0; i < 8; i++)
    cvt_f32_to_f16<<<1024, 256, 0, stream>>>(Wf[i], Wh[i], E * E / 4);

  dim3 ggrid(E / 128, MTOT / 128); // 8 x 32
  dim3 agrid(SEQ / 64, H, BATCH);  // 32 x 16 x 2

  // ---- t2s: Q from spat, K/V from temp, residual spat -> out[second half]
  gemm_bias_kernel<true ><<<ggrid, 256, 0, stream>>>(spat_h, Wh[0], bf[0], Qh, MTOT, E, E);
  gemm_bias_kernel<true ><<<ggrid, 256, 0, stream>>>(temp_h, Wh[1], bf[1], Kh, MTOT, E, E);
  gemm_bias_kernel<true ><<<ggrid, 256, 0, stream>>>(temp_h, Wh[2], bf[2], Vh, MTOT, E, E);
  attn_kernel<<<agrid, 128, 0, stream>>>(Qh, Kh, Vh, Oh);
  gemm_bias_kernel<false><<<ggrid, 256, 0, stream>>>(Oh, Wh[3], bf[3], Yb, MTOT, E, E);
  add_ln_kernel<<<MTOT, 256, 0, stream>>>(spat, Yb, ln_g, ln_b, out + (size_t)MTOT * E);

  // ---- s2t: Q from temp, K/V from spat, residual temp -> out[first half]
  gemm_bias_kernel<true ><<<ggrid, 256, 0, stream>>>(temp_h, Wh[4], bf[4], Qh, MTOT, E, E);
  gemm_bias_kernel<true ><<<ggrid, 256, 0, stream>>>(spat_h, Wh[5], bf[5], Kh, MTOT, E, E);
  gemm_bias_kernel<true ><<<ggrid, 256, 0, stream>>>(spat_h, Wh[6], bf[6], Vh, MTOT, E, E);
  attn_kernel<<<agrid, 128, 0, stream>>>(Qh, Kh, Vh, Oh);
  gemm_bias_kernel<false><<<ggrid, 256, 0, stream>>>(Oh, Wh[7], bf[7], Yb, MTOT, E, E);
  add_ln_kernel<<<MTOT, 256, 0, stream>>>(temp, Yb, ln_g, ln_b, out);
}
